// FusionBlock_44538810859955
// MI455X (gfx1250) — compile-verified, matrix operand order checked by reading, not run on hardware
//
#include <hip/hip_runtime.h>

typedef __bf16 bf16_t;
typedef __attribute__((ext_vector_type(16))) __bf16 v16bf;
typedef __attribute__((ext_vector_type(8)))  __bf16 v8bf;
typedef __attribute__((ext_vector_type(8)))  float  v8f;

#define B2   2
#define LSP  16384      // 128*128
#define MSP  32768      // B2*LSP

__device__ __forceinline__ float rcp_fast(float x){ return __builtin_amdgcn_rcpf(x); }
__device__ __forceinline__ float rsq_fast(float x){ return __builtin_amdgcn_rsqf(x); }
__device__ __forceinline__ float lrelu_f(float x){ return x > 0.f ? x : 0.2f*x; }
__device__ __forceinline__ float sigm_f (float x){ return rcp_fast(1.f + __expf(-x)); }
__device__ __forceinline__ float silu_f (float x){ return x*sigm_f(x); }
__device__ __forceinline__ float splus_f(float x){ return x > 20.f ? x : log1pf(__expf(x)); }
__device__ __forceinline__ float gelu_f (float x){ return 0.5f*x*(1.f+erff(x*0.70710678f)); }

// ---------------- WMMA helpers (16x16 tile, K step 32, bf16 A/B, f32 acc) ---------
// A fragment (16-bit, 16x32): lane = half*16 + m; elements j<8 -> k = k0+half*8+j,
// j>=8 -> k = k0+16+half*8+(j-8). B fragment mirrors with n = lane%16.
__device__ __forceinline__ v16bf frag_zero(){
  v16bf r;
#pragma unroll
  for (int i = 0; i < 16; ++i) r[i] = (bf16_t)0.f;
  return r;
}
__device__ __forceinline__ v16bf load_frag(const bf16_t* p0, int lane){
  const bf16_t* p = p0 + ((lane >> 4) << 3);
  v8bf lo = *reinterpret_cast<const v8bf*>(p);
  v8bf hi = *reinterpret_cast<const v8bf*>(p + 16);
  v16bf r;
#pragma unroll
  for (int i = 0; i < 8; ++i) { r[i] = lo[i]; r[i+8] = hi[i]; }
  return r;
}
__device__ __forceinline__ v8f wmma_bf16(v16bf a, v16bf b, v8f c){
  return __builtin_amdgcn_wmma_f32_16x16x32_bf16(false, a, false, b, (short)0, c, false, false);
}
// One A fragment reused against NT consecutive 16-wide B tiles (16 x NT*16 output).
template <int NT>
__device__ __forceinline__ void gemm_tile_nt(const bf16_t* A, int lda, const bf16_t* Bt, int ldb,
                                             int K, int m0, int n0, int lane, v8f (&acc)[NT]){
  int r = lane & 15;
  const bf16_t* arow = A  + (size_t)(m0 + r) * lda;
  const bf16_t* brow = Bt + (size_t)(n0 + r) * ldb;
  for (int k0 = 0; k0 < K; k0 += 32){
    v16bf a = load_frag(arow + k0, lane);
#pragma unroll
    for (int j = 0; j < NT; ++j){
      v16bf b = load_frag(brow + (size_t)j*16*ldb + k0, lane);
      acc[j] = wmma_bf16(a, b, acc[j]);
    }
  }
}
// D layout: acc[j][r2] = D[m0 + r2 + 8*(lane>>4)][n0 + j*16 + lane%16]

// ---------------- weight prep kernels -------------------------------------------
__global__ void k_cvt_copy(const float* s, bf16_t* d, int n){
  int i = blockIdx.x*256 + threadIdx.x; if (i < n) d[i] = (bf16_t)s[i];
}
__global__ void k_cvt_transpose(const float* s, bf16_t* d, int R, int C){
  int i = blockIdx.x*256 + threadIdx.x;
  if (i < R*C){ int r = i / C, c = i % C; d[(size_t)c*R + r] = (bf16_t)s[i]; }
}
__global__ void k_cvt_xprojT(const float* s, bf16_t* d){   // [4][36][128] -> [4][48][128] pad
  int i = blockIdx.x*256 + threadIdx.x;
  if (i < 4*48*128){
    int k = i / (48*128); int n = (i/128) % 48; int c = i % 128;
    d[i] = (bf16_t)((n < 36) ? s[(k*36+n)*128 + c] : 0.f);
  }
}
__global__ void k_cvt_adjT(const float* s, bf16_t* d){     // [oc][ic][3][3] -> [oc][9][ic]
  int i = blockIdx.x*256 + threadIdx.x;
  if (i < 128*9*128){
    int oc = i/(9*128); int t = (i/128)%9; int ic = i%128;
    d[i] = (bf16_t)s[((oc*128+ic)*9) + t];
  }
}
__global__ void k_scale_projT(const bf16_t* w, const float* s, bf16_t* d){ // per-batch SE scale
  int i = blockIdx.x*256 + threadIdx.x;
  if (i < B2*64*256){ int b = i/(64*256); int c = i%256;
    d[i] = (bf16_t)((float)w[i % (64*256)] * s[b*256 + c]); }
}
__global__ void k_scale_adjT(const bf16_t* w, const float* s, bf16_t* d){ // per-batch top-CA scale
  int i = blockIdx.x*256 + threadIdx.x;
  if (i < B2*128*9*128){ int b = i/(128*9*128); int ic = i%128;
    d[i] = (bf16_t)((float)w[i % (128*9*128)] * s[b*128 + ic]); }
}

// ---------------- LN1 + activation prep (NCHW -> NHWC bf16) ----------------------
__global__ void k_ln1_prep(const float* x, const float* g, const float* be,
                           bf16_t* yln, bf16_t* ax1){
  int m = blockIdx.x; int c = threadIdx.x;           // 32768 x 64
  int b = m >> 14; int hw = m & 16383;
  float v1 = x[(((size_t)(b*128 + c))      << 14) + hw];
  float v2 = x[(((size_t)(b*128 + 64 + c)) << 14) + hw];
  ax1[(size_t)m*64 + c] = (bf16_t)v1;
  __shared__ float red[64];
  red[c] = v2; __syncthreads();
  for (int s = 32; s > 0; s >>= 1){ if (c < s) red[c] += red[c+s]; __syncthreads(); }
  float mean = red[0] * (1.f/64.f); __syncthreads();
  float dv = v2 - mean; red[c] = dv*dv; __syncthreads();
  for (int s = 32; s > 0; s >>= 1){ if (c < s) red[c] += red[c+s]; __syncthreads(); }
  float rstd = rsq_fast(red[0]*(1.f/64.f) + 1e-5f);
  yln[(size_t)m*64 + c] = (bf16_t)(dv*rstd*g[c] + be[c]);
}

// ---------------- GEMM kernels (one wave = 16 x NT*16 output) --------------------
__global__ void g_inproj(const bf16_t* A, const bf16_t* Bt, bf16_t* xi, bf16_t* zs){
  int lane = threadIdx.x & 31;
  int wave = (blockIdx.x*blockDim.x + threadIdx.x) >> 5;   // 8192 waves
  int mt = wave >> 2, ng = wave & 3;
  v8f acc[4] = {};
  gemm_tile_nt<4>(A, 64, Bt, 64, 64, mt*16, ng*64, lane, acc);
  int l15 = lane & 15; int h8 = (lane >> 4) << 3;
#pragma unroll
  for (int j = 0; j < 4; ++j){
    int tb = ng*64 + j*16;          // tile base (wave-uniform)
    int n = tb + l15;
#pragma unroll
    for (int r2 = 0; r2 < 8; ++r2){
      int m = mt*16 + r2 + h8; float v = acc[j][r2];
      if (tb < 128) xi[(size_t)m*128 + n] = (bf16_t)v;
      else          zs[(size_t)m*128 + (n-128)] = (bf16_t)silu_f(v);
    }
  }
}
__global__ void g_exp(const bf16_t* A, const bf16_t* Bt, const float* bias,
                      const float* g, const float* be, const float* mn, const float* vr,
                      bf16_t* h1){
  int lane = threadIdx.x & 31;
  int wave = (blockIdx.x*blockDim.x + threadIdx.x) >> 5;   // 8192 waves
  int mt = wave >> 2, ng = wave & 3;
  v8f acc[4] = {};
  gemm_tile_nt<4>(A, 64, Bt, 64, 64, mt*16, ng*64, lane, acc);
  int l15 = lane & 15; int h8 = (lane >> 4) << 3;
#pragma unroll
  for (int j = 0; j < 4; ++j){
    int n = ng*64 + j*16 + l15;
    float sc = g[n]*rsq_fast(vr[n]+1e-5f);
#pragma unroll
    for (int r2 = 0; r2 < 8; ++r2){
      int m = mt*16 + r2 + h8;
      float v = acc[j][r2] + bias[n];
      v = (v - mn[n])*sc + be[n];
      h1[(size_t)m*256 + n] = (bf16_t)lrelu_f(v);
    }
  }
}
__global__ void g_xproj(const bf16_t* xc, const bf16_t* xsT1, const bf16_t* Bt4, bf16_t* xdbl){
  int k = blockIdx.y;
  int lane = threadIdx.x & 31;
  int wave = (blockIdx.x*blockDim.x + threadIdx.x) >> 5;   // 2048 waves (mt)
  int mt = wave;
  const bf16_t* A = (k & 1) ? xsT1 : xc;
  v8f acc[3] = {};
  gemm_tile_nt<3>(A, 128, Bt4 + (size_t)k*48*128, 128, 128, mt*16, 0, lane, acc);
  int l15 = lane & 15; int h8 = (lane >> 4) << 3;
#pragma unroll
  for (int j = 0; j < 3; ++j){
    int n = j*16 + l15;
#pragma unroll
    for (int r2 = 0; r2 < 8; ++r2){
      int m = mt*16 + r2 + h8; int b = m >> 14; int l = m & 16383;
      xdbl[(((size_t)(b*4 + k)*LSP + l)*48) + n] = (bf16_t)acc[j][r2];
    }
  }
}
__global__ void g_outproj(const bf16_t* A, const bf16_t* Bt, const float* x,
                          const float* skip1, float* xAfter){
  int lane = threadIdx.x & 31;
  int wave = (blockIdx.x*blockDim.x + threadIdx.x) >> 5;   // 2048 waves
  int mt = wave;
  v8f acc[4] = {};
  gemm_tile_nt<4>(A, 128, Bt, 128, 128, mt*16, 0, lane, acc);
  int l15 = lane & 15; int h8 = (lane >> 4) << 3;
#pragma unroll
  for (int j = 0; j < 4; ++j){
    int n = j*16 + l15;
#pragma unroll
    for (int r2 = 0; r2 < 8; ++r2){
      int m = mt*16 + r2 + h8; int b = m >> 14; int l = m & 16383;
      float x2v = x[(((size_t)(b*128 + 64 + n)) << 14) + l];
      xAfter[(size_t)m*64 + n] = x2v*skip1[n] + acc[j][r2];
    }
  }
}
__global__ void g_proj(const bf16_t* A, const bf16_t* BtB, const float* bias,
                       const float* g, const float* be, const float* mn, const float* vr,
                       const float* x, bf16_t* combs){
  int lane = threadIdx.x & 31;
  int wave = (blockIdx.x*blockDim.x + threadIdx.x) >> 5;   // 2048 waves
  int mt = wave;
  int b = (mt*16) >> 14;
  v8f acc[4] = {};
  gemm_tile_nt<4>(A, 256, BtB + (size_t)b*64*256, 256, 256, mt*16, 0, lane, acc);
  int l15 = lane & 15; int h8 = (lane >> 4) << 3;
#pragma unroll
  for (int j = 0; j < 4; ++j){
    int n = j*16 + l15;
    float sc = g[n]*rsq_fast(vr[n]+1e-5f);
#pragma unroll
    for (int r2 = 0; r2 < 8; ++r2){
      int m = mt*16 + r2 + h8; int l = m & 16383;
      float v = acc[j][r2] + bias[n];
      v = lrelu_f((v - mn[n])*sc + be[n]);
      v += x[(((size_t)(b*128 + n)) << 14) + l];
      combs[(size_t)m*128 + n] = (bf16_t)v;
    }
  }
}
// 3x3 conv as tap-looped WMMA GEMM (K = 9*128), s_top folded into weights.
__global__ void g_adj(const bf16_t* combs, const bf16_t* adjTb, const float* adjb,
                      const float* xin, float* out){
  int lane = threadIdx.x & 31;
  int wave = (blockIdx.x*blockDim.x + threadIdx.x) >> 5;   // 4096 waves
  int mt = wave >> 1, ng = wave & 1;
  int m0 = mt << 4, n0 = ng << 6;
  int r = lane & 15;
  int m = m0 + r;
  int b = m >> 14; int hw = m & 16383; int h = hw >> 7; int w = hw & 127;
  const bf16_t* Bt = adjTb + (size_t)b*128*1152 + (size_t)(n0 + r)*1152;
  v8f acc[4] = {};
  for (int t = 0; t < 9; ++t){
    int dy = t/3 - 1, dx = t%3 - 1;
    int hh = h + dy, ww = w + dx;
    bool ok = (hh >= 0) && (hh < 128) && (ww >= 0) && (ww < 128);
    int hc = min(max(hh, 0), 127), wc = min(max(ww, 0), 127);
    const bf16_t* arow = combs + (((size_t)(b << 14) + (hc << 7) + wc) << 7);
    const bf16_t* brow = Bt + t*128;
    for (int k0 = 0; k0 < 128; k0 += 32){
      v16bf a = ok ? load_frag(arow + k0, lane) : frag_zero();
#pragma unroll
      for (int j = 0; j < 4; ++j){
        v16bf bb = load_frag(brow + (size_t)j*16*1152 + k0, lane);
        acc[j] = wmma_bf16(a, bb, acc[j]);
      }
    }
  }
  int l15 = lane & 15; int h8 = (lane >> 4) << 3;
#pragma unroll
  for (int j = 0; j < 4; ++j){
    int n = n0 + j*16 + l15;
#pragma unroll
    for (int r2 = 0; r2 < 8; ++r2){
      int mm = m0 + r2 + h8; int l = mm & 16383;
      size_t oi = (((size_t)(b*128 + n)) << 14) + l;
      out[oi] = acc[j][r2] + adjb[n] + xin[oi];
    }
  }
}

// ---------------- depthwise convs (NHWC) -----------------------------------------
__global__ void k_dw_mb(const bf16_t* in, const float* wt, const float* bias,
                        const float* g, const float* be, const float* mn, const float* vr,
                        bf16_t* out){
  size_t i = (size_t)blockIdx.x*256 + threadIdx.x;   // 32768*256
  int c = i & 255; size_t m = i >> 8;
  int b = (int)(m >> 14); int hw = (int)(m & 16383); int h = hw >> 7, w = hw & 127;
  float a = bias[c];
#pragma unroll
  for (int t = 0; t < 9; ++t){
    int hh = h + t/3 - 1, ww = w + t%3 - 1;
    if (hh >= 0 && hh < 128 && ww >= 0 && ww < 128)
      a += (float)in[(((size_t)(b << 14) + (hh << 7) + ww) << 8) + c] * wt[c*9 + t];
  }
  a = (a - mn[c])*(g[c]*rsq_fast(vr[c]+1e-5f)) + be[c];
  out[i] = (bf16_t)lrelu_f(a);
}
__global__ void k_dw_vss(const bf16_t* in, const float* wt, const float* bias, bf16_t* out){
  size_t i = (size_t)blockIdx.x*256 + threadIdx.x;   // 32768*128
  int c = i & 127; size_t m = i >> 7;
  int b = (int)(m >> 14); int hw = (int)(m & 16383); int h = hw >> 7, w = hw & 127;
  float a = bias[c];
#pragma unroll
  for (int t = 0; t < 9; ++t){
    int hh = h + t/3 - 1, ww = w + t%3 - 1;
    if (hh >= 0 && hh < 128 && ww >= 0 && ww < 128)
      a += (float)in[(((size_t)(b << 14) + (hh << 7) + ww) << 7) + c] * wt[c*9 + t];
  }
  out[i] = (bf16_t)silu_f(a);
}
__global__ void k_xsT1(const bf16_t* xc, bf16_t* xsT1){   // col-major scan order copy
  size_t i = (size_t)blockIdx.x*256 + threadIdx.x;        // 32768*128
  int d = i & 127; size_t m = i >> 7;
  int b = (int)(m >> 14); int hw = (int)(m & 16383); int h = hw >> 7, w = hw & 127;
  xsT1[(((size_t)(b << 14)) + (w << 7) + h)*128 + d] = xc[i];
}

// ---------------- chunked selective scan -----------------------------------------
__global__ void k_scan_p1(const bf16_t* xdbl, const bf16_t* xsT0, const bf16_t* xsT1,
                          const float* dtw, const float* dtb, const float* Alog,
                          float* chunkP, float* chunkQ){
  int chunk = blockIdx.x; int bk = blockIdx.y;
  int b = bk >> 2, k = bk & 3; int d = threadIdx.x;   // 128
  float Ad[16];
#pragma unroll
  for (int n = 0; n < 16; ++n) Ad[n] = -__expf(Alog[(k*128 + d)*16 + n]);
  float w0 = dtw[(k*128+d)*4+0], w1 = dtw[(k*128+d)*4+1];
  float w2 = dtw[(k*128+d)*4+2], w3 = dtw[(k*128+d)*4+3];
  float bb = dtb[k*128 + d];
  float h[16], P[16];
#pragma unroll
  for (int n = 0; n < 16; ++n){ h[n] = 0.f; P[n] = 1.f; }
  const bf16_t* xsT = (k & 1) ? xsT1 : xsT0;
  __shared__ float rowBuf[48];
  for (int s = 0; s < 128; ++s){
    int l = chunk*128 + s;
    int row = (k < 2) ? l : (LSP - 1 - l);
    __syncthreads();
    if (d < 48) rowBuf[d] = (float)xdbl[((size_t)bk*LSP + row)*48 + d];
    __syncthreads();
    float dt = splus_f(w0*rowBuf[0] + w1*rowBuf[1] + w2*rowBuf[2] + w3*rowBuf[3] + bb);
    float xv = (float)xsT[((size_t)b*LSP + row)*128 + d];
    float dtx = dt*xv;
#pragma unroll
    for (int n = 0; n < 16; ++n){
      float dA = __expf(dt*Ad[n]);
      h[n] = dA*h[n] + dtx*rowBuf[4 + n];
      P[n] *= dA;
    }
  }
  size_t o = (((size_t)bk*128 + chunk)*128 + d)*16;
#pragma unroll
  for (int n = 0; n < 16; ++n){ chunkP[o + n] = P[n]; chunkQ[o + n] = h[n]; }
}
__global__ void k_scan_p2(const float* P, const float* Q, float* hIn){
  int bk = blockIdx.x; int d = threadIdx.x;
  float h[16];
#pragma unroll
  for (int n = 0; n < 16; ++n) h[n] = 0.f;
  for (int c = 0; c < 128; ++c){
    size_t o = (((size_t)bk*128 + c)*128 + d)*16;
#pragma unroll
    for (int n = 0; n < 16; ++n) hIn[o + n] = h[n];
#pragma unroll
    for (int n = 0; n < 16; ++n) h[n] = P[o + n]*h[n] + Q[o + n];
  }
}
__global__ void k_scan_p3(const bf16_t* xdbl, const bf16_t* xsT0, const bf16_t* xsT1,
                          const float* dtw, const float* dtb, const float* Alog,
                          const float* Ds, const float* hIn, bf16_t* ysStore){
  int chunk = blockIdx.x; int bk = blockIdx.y;
  int b = bk >> 2, k = bk & 3; int d = threadIdx.x;
  float Ad[16];
#pragma unroll
  for (int n = 0; n < 16; ++n) Ad[n] = -__expf(Alog[(k*128 + d)*16 + n]);
  float w0 = dtw[(k*128+d)*4+0], w1 = dtw[(k*128+d)*4+1];
  float w2 = dtw[(k*128+d)*4+2], w3 = dtw[(k*128+d)*4+3];
  float bb = dtb[k*128 + d];
  float Dv = Ds[k*128 + d];
  float h[16];
  size_t ho = (((size_t)bk*128 + chunk)*128 + d)*16;
#pragma unroll
  for (int n = 0; n < 16; ++n) h[n] = hIn[ho + n];
  const bf16_t* xsT = (k & 1) ? xsT1 : xsT0;
  __shared__ float rowBuf[48];
  for (int s = 0; s < 128; ++s){
    int l = chunk*128 + s;
    int row = (k < 2) ? l : (LSP - 1 - l);
    __syncthreads();
    if (d < 48) rowBuf[d] = (float)xdbl[((size_t)bk*LSP + row)*48 + d];
    __syncthreads();
    float dt = splus_f(w0*rowBuf[0] + w1*rowBuf[1] + w2*rowBuf[2] + w3*rowBuf[3] + bb);
    float xv = (float)xsT[((size_t)b*LSP + row)*128 + d];
    float dtx = dt*xv;
    float y = 0.f;
#pragma unroll
    for (int n = 0; n < 16; ++n){
      float dA = __expf(dt*Ad[n]);
      h[n] = dA*h[n] + dtx*rowBuf[4 + n];
      y += h[n]*rowBuf[20 + n];
    }
    y += Dv*xv;
    ysStore[((size_t)bk*LSP + row)*128 + d] = (bf16_t)y;
  }
}

// ---------------- combine 4 scan dirs + out_norm + z gating ----------------------
__global__ void k_combine(const bf16_t* ys, const float* g, const float* be,
                          const bf16_t* zs, bf16_t* yg){
  int m = blockIdx.x; int d = threadIdx.x;            // 32768 x 128
  int b = m >> 14; int l = m & 16383; int h = l >> 7, w = l & 127;
  int mt = (w << 7) + h;
  size_t bb = (size_t)b*4*((size_t)LSP*128);
  float y = (float)ys[bb + 0*(size_t)LSP*128 + (size_t)l *128 + d]
          + (float)ys[bb + 2*(size_t)LSP*128 + (size_t)l *128 + d]
          + (float)ys[bb + 1*(size_t)LSP*128 + (size_t)mt*128 + d]
          + (float)ys[bb + 3*(size_t)LSP*128 + (size_t)mt*128 + d];
  __shared__ float red[128];
  red[d] = y; __syncthreads();
  for (int s = 64; s > 0; s >>= 1){ if (d < s) red[d] += red[d+s]; __syncthreads(); }
  float mean = red[0]*(1.f/128.f); __syncthreads();
  float dv = y - mean; red[d] = dv*dv; __syncthreads();
  for (int s = 64; s > 0; s >>= 1){ if (d < s) red[d] += red[d+s]; __syncthreads(); }
  float rstd = rsq_fast(red[0]*(1.f/128.f) + 1e-5f);
  float v = dv*rstd*g[d] + be[d];
  yg[(size_t)m*128 + d] = (bf16_t)(v * (float)zs[(size_t)m*128 + d]);
}

// ---------------- LN2 / CAB branch -----------------------------------------------
__global__ void k_ln2(const float* xA, const float* g, const float* be, float* out){
  int m = blockIdx.x; int c = threadIdx.x;            // 32768 x 64
  float v = xA[(size_t)m*64 + c];
  __shared__ float red[64];
  red[c] = v; __syncthreads();
  for (int s = 32; s > 0; s >>= 1){ if (c < s) red[c] += red[c+s]; __syncthreads(); }
  float mean = red[0]*(1.f/64.f); __syncthreads();
  float dv = v - mean; red[c] = dv*dv; __syncthreads();
  for (int s = 32; s > 0; s >>= 1){ if (c < s) red[c] += red[c+s]; __syncthreads(); }
  float rstd = rsq_fast(red[0]*(1.f/64.f) + 1e-5f);
  out[(size_t)m*64 + c] = dv*rstd*g[c] + be[c];
}
__global__ void k_cab1(const float* in, const float* wt, const float* bias, float* c1){
  size_t i = (size_t)blockIdx.x*256 + threadIdx.x;
  if (i >= (size_t)MSP*21) return;
  int oc = (int)(i % 21); size_t m = i / 21;
  int b = (int)(m >> 14); int hw = (int)(m & 16383); int h = hw >> 7, w = hw & 127;
  float a = bias[oc];
  for (int t = 0; t < 9; ++t){
    int hh = h + t/3 - 1, ww = w + t%3 - 1;
    if (hh < 0 || hh > 127 || ww < 0 || ww > 127) continue;
    const float* row = in + (((size_t)(b << 14) + (hh << 7) + ww)*64);
    for (int ic = 0; ic < 64; ++ic) a += row[ic]*wt[(oc*64 + ic)*9 + t];
  }
  c1[m*24 + oc] = gelu_f(a);
}
__global__ void k_cab2(const float* in, const float* wt, const float* bias, float* c2){
  size_t i = (size_t)blockIdx.x*256 + threadIdx.x;    // 32768*64
  int oc = (int)(i & 63); size_t m = i >> 6;
  int b = (int)(m >> 14); int hw = (int)(m & 16383); int h = hw >> 7, w = hw & 127;
  float a = bias[oc];
  for (int t = 0; t < 9; ++t){
    int hh = h + t/3 - 1, ww = w + t%3 - 1;
    if (hh < 0 || hh > 127 || ww < 0 || ww > 127) continue;
    const float* row = in + (((size_t)(b << 14) + (hh << 7) + ww)*24);
    for (int ic = 0; ic < 21; ++ic) a += row[ic]*wt[(oc*21 + ic)*9 + t];
  }
  c2[i] = a;
}
__global__ void k_vssout(const float* xA, const float* skip2, const float* c2,
                         const float* scab, bf16_t* combs){
  size_t i = (size_t)blockIdx.x*256 + threadIdx.x;    // 32768*64
  int c = (int)(i & 63); size_t m = i >> 6; int b = (int)(m >> 14);
  float v = xA[i]*skip2[c] + c2[i]*scab[b*64 + c];
  combs[m*128 + 64 + c] = (bf16_t)v;
}

// ---------------- pooling + tiny FC (channel attention / SE) ---------------------
template <typename T>
__global__ void k_pool_part(const T* src, float* partial, int C){
  int b = blockIdx.y, slice = blockIdx.x, c = threadIdx.x;
  float a = 0.f;
  for (int i = 0; i < 256; ++i){
    int l = slice*256 + i;
    a += (float)src[((size_t)(b*LSP + l))*C + c];
  }
  partial[(size_t)(b*64 + slice)*C + c] = a;
}
__global__ void k_pool_fin(const float* partial, float* out, int C){
  int b = blockIdx.x, c = threadIdx.x;
  float a = 0.f;
  for (int s = 0; s < 64; ++s) a += partial[(size_t)(b*64 + s)*C + c];
  out[b*C + c] = a * (1.f/(float)LSP);
}
__global__ void k_fc(const float* in, const float* w, const float* bias,
                     float* out, int Cin, int Cout, int act){
  int t = threadIdx.x; int b = t / Cout; int o = t % Cout;
  if (b >= B2) return;
  float a = bias ? bias[o] : 0.f;
  for (int i = 0; i < Cin; ++i) a += in[b*Cin + i]*w[o*Cin + i];
  if (act == 0) a = lrelu_f(a);
  else if (act == 1) a = fmaxf(a, 0.f);
  else a = sigm_f(a);
  out[b*Cout + o] = a;
}

// ================================================================================
extern "C" void kernel_launch(void* const* d_in, const int* in_sizes, int n_in,
                              void* d_out, int out_size, void* d_ws, size_t ws_size,
                              hipStream_t stream){
  (void)in_sizes; (void)n_in; (void)out_size; (void)ws_size;
  const float* X     = (const float*)d_in[0];
  // mbconv params
  const float* EXPW  = (const float*)d_in[1];  const float* EXPB = (const float*)d_in[2];
  const float* BN1G  = (const float*)d_in[3];  const float* BN1B = (const float*)d_in[4];
  const float* BN1M  = (const float*)d_in[5];  const float* BN1V = (const float*)d_in[6];
  const float* DWW   = (const float*)d_in[7];  const float* DWB  = (const float*)d_in[8];
  const float* BN2G  = (const float*)d_in[9];  const float* BN2B = (const float*)d_in[10];
  const float* BN2M  = (const float*)d_in[11]; const float* BN2V = (const float*)d_in[12];
  const float* SEW1  = (const float*)d_in[13]; const float* SEB1 = (const float*)d_in[14];
  const float* SEW2  = (const float*)d_in[15]; const float* SEB2 = (const float*)d_in[16];
  const float* PRW   = (const float*)d_in[17]; const float* PRB  = (const float*)d_in[18];
  const float* BN3G  = (const float*)d_in[19]; const float* BN3B = (const float*)d_in[20];
  const float* BN3M  = (const float*)d_in[21]; const float* BN3V = (const float*)d_in[22];
  // vss params
  const float* LN1G  = (const float*)d_in[23]; const float* LN1B = (const float*)d_in[24];
  const float* INPW  = (const float*)d_in[25];
  const float* CONVW = (const float*)d_in[26]; const float* CONVB= (const float*)d_in[27];
  const float* XPW   = (const float*)d_in[28];
  const float* DTW   = (const float*)d_in[29]; const float* DTB  = (const float*)d_in[30];
  const float* ALOG  = (const float*)d_in[31]; const float* DS   = (const float*)d_in[32];
  const float* ONG   = (const float*)d_in[33]; const float* ONB  = (const float*)d_in[34];
  const float* OUTPW = (const float*)d_in[35]; const float* SKIP1= (const float*)d_in[36];
  const float* LN2G  = (const float*)d_in[37]; const float* LN2B = (const float*)d_in[38];
  const float* CABW1 = (const float*)d_in[39]; const float* CABB1= (const float*)d_in[40];
  const float* CABW2 = (const float*)d_in[41]; const float* CABB2= (const float*)d_in[42];
  const float* CAW1  = (const float*)d_in[43]; const float* CAB1 = (const float*)d_in[44];
  const float* CAW2  = (const float*)d_in[45]; const float* CAB2 = (const float*)d_in[46];
  const float* SKIP2 = (const float*)d_in[47];
  // top params
  const float* TCAW1 = (const float*)d_in[48]; const float* TCAW2= (const float*)d_in[49];
  const float* ADJW  = (const float*)d_in[50]; const float* ADJB = (const float*)d_in[51];
  float* OUT = (float*)d_out;

  char* base = (char*)d_ws; size_t off = 0;
  auto AL = [&](size_t bytes) -> char* {
    char* p = base + off; off = (off + bytes + 255) & ~(size_t)255; return p;
  };
  // bf16 weight pool
  bf16_t* inprojT = (bf16_t*)AL(256*64*2);
  bf16_t* xprojT  = (bf16_t*)AL(4*48*128*2);
  bf16_t* outprojT= (bf16_t*)AL(64*128*2);
  bf16_t* expT    = (bf16_t*)AL(256*64*2);
  bf16_t* projT   = (bf16_t*)AL(64*256*2);
  bf16_t* projTb  = (bf16_t*)AL((size_t)B2*64*256*2);
  bf16_t* adjT    = (bf16_t*)AL((size_t)128*9*128*2);
  bf16_t* adjTb   = (bf16_t*)AL((size_t)B2*128*9*128*2);
  // activations
  bf16_t* yln   = (bf16_t*)AL((size_t)MSP*64*2);
  bf16_t* ax1   = (bf16_t*)AL((size_t)MSP*64*2);
  bf16_t* h1    = (bf16_t*)AL((size_t)MSP*256*2);
  bf16_t* h2    = (bf16_t*)AL((size_t)MSP*256*2);
  bf16_t* xi    = (bf16_t*)AL((size_t)MSP*128*2);
  bf16_t* zsb   = (bf16_t*)AL((size_t)MSP*128*2);
  bf16_t* xc    = (bf16_t*)AL((size_t)MSP*128*2);   // == xsT0
  bf16_t* xsT1  = (bf16_t*)AL((size_t)MSP*128*2);
  bf16_t* xdbl  = (bf16_t*)AL((size_t)B2*4*LSP*48*2);
  float*  chP   = (float*)AL((size_t)B2*4*128*128*16*4);
  float*  chQ   = (float*)AL((size_t)B2*4*128*128*16*4);
  float*  hIn   = (float*)AL((size_t)B2*4*128*128*16*4);
  bf16_t* ysS   = (bf16_t*)AL((size_t)B2*4*LSP*128*2);
  bf16_t* yg    = (bf16_t*)AL((size_t)MSP*128*2);
  float*  xAfter= (float*)AL((size_t)MSP*64*4);
  float*  ln2o  = (float*)AL((size_t)MSP*64*4);
  float*  c1b   = (float*)AL((size_t)MSP*24*4);
  float*  c2b   = (float*)AL((size_t)MSP*64*4);
  bf16_t* combs = (bf16_t*)AL((size_t)MSP*128*2);
  // pooling / attention scratch
  float* part   = (float*)AL((size_t)B2*64*256*4);
  float* poolSE = (float*)AL(B2*256*4);
  float* seMid  = (float*)AL(B2*64*4);
  float* sSE    = (float*)AL(B2*256*4);
  float* poolCab= (float*)AL(B2*64*4);
  float* cabMid = (float*)AL(B2*2*4);
  float* sCab   = (float*)AL(B2*64*4);
  float* poolTop= (float*)AL(B2*128*4);
  float* topMid = (float*)AL(B2*8*4);
  float* sTop   = (float*)AL(B2*128*4);

  // ---- weight prep ----
  k_cvt_transpose<<<(64*256+255)/256, 256, 0, stream>>>(INPW, inprojT, 64, 256);
  k_cvt_xprojT   <<<(4*48*128+255)/256, 256, 0, stream>>>(XPW, xprojT);
  k_cvt_transpose<<<(128*64+255)/256, 256, 0, stream>>>(OUTPW, outprojT, 128, 64);
  k_cvt_copy     <<<(256*64+255)/256, 256, 0, stream>>>(EXPW, expT, 256*64);
  k_cvt_copy     <<<(64*256+255)/256, 256, 0, stream>>>(PRW, projT, 64*256);
  k_cvt_adjT     <<<(128*9*128+255)/256, 256, 0, stream>>>(ADJW, adjT);

  // ---- input prep: split channels, LN1 ----
  k_ln1_prep<<<MSP, 64, 0, stream>>>(X, LN1G, LN1B, yln, ax1);

  // ---- MBConv branch ----
  g_exp<<<2048, 128, 0, stream>>>(ax1, expT, EXPB, BN1G, BN1B, BN1M, BN1V, h1);
  k_dw_mb<<<MSP, 256, 0, stream>>>(h1, DWW, DWB, BN2G, BN2B, BN2M, BN2V, h2);
  k_pool_part<bf16_t><<<dim3(64, B2), 256, 0, stream>>>(h2, part, 256);
  k_pool_fin<<<B2, 256, 0, stream>>>(part, poolSE, 256);
  k_fc<<<1, B2*64, 0, stream>>>(poolSE, SEW1, SEB1, seMid, 256, 64, 0);
  k_fc<<<1, B2*256, 0, stream>>>(seMid, SEW2, SEB2, sSE, 64, 256, 2);
  k_scale_projT<<<(B2*64*256+255)/256, 256, 0, stream>>>(projT, sSE, projTb);
  g_proj<<<512, 128, 0, stream>>>(h2, projTb, PRB, BN3G, BN3B, BN3M, BN3V, X, combs);

  // ---- VSS branch ----
  g_inproj<<<2048, 128, 0, stream>>>(yln, inprojT, xi, zsb);
  k_dw_vss<<<MSP/2, 256, 0, stream>>>(xi, CONVW, CONVB, xc);
  k_xsT1<<<MSP/2, 256, 0, stream>>>(xc, xsT1);
  g_xproj<<<dim3(512, 4), 128, 0, stream>>>(xc, xsT1, xprojT, xdbl);
  k_scan_p1<<<dim3(128, B2*4), 128, 0, stream>>>(xdbl, xc, xsT1, DTW, DTB, ALOG, chP, chQ);
  k_scan_p2<<<B2*4, 128, 0, stream>>>(chP, chQ, hIn);
  k_scan_p3<<<dim3(128, B2*4), 128, 0, stream>>>(xdbl, xc, xsT1, DTW, DTB, ALOG, DS, hIn, ysS);
  k_combine<<<MSP, 128, 0, stream>>>(ysS, ONG, ONB, zsb, yg);
  g_outproj<<<512, 128, 0, stream>>>(yg, outprojT, X, SKIP1, xAfter);

  // ---- CAB + channel attention ----
  k_ln2<<<MSP, 64, 0, stream>>>(xAfter, LN2G, LN2B, ln2o);
  k_cab1<<<(int)(((size_t)MSP*21 + 255)/256), 256, 0, stream>>>(ln2o, CABW1, CABB1, c1b);
  k_cab2<<<MSP/4, 256, 0, stream>>>(c1b, CABW2, CABB2, c2b);
  k_pool_part<float><<<dim3(64, B2), 64, 0, stream>>>(c2b, part, 64);
  k_pool_fin<<<B2, 64, 0, stream>>>(part, poolCab, 64);
  k_fc<<<1, B2*2, 0, stream>>>(poolCab, CAW1, CAB1, cabMid, 64, 2, 1);
  k_fc<<<1, B2*64, 0, stream>>>(cabMid, CAW2, CAB2, sCab, 2, 64, 2);
  k_vssout<<<MSP/4, 256, 0, stream>>>(xAfter, SKIP2, c2b, sCab, combs);

  // ---- top fusion: channel attention folded into 3x3 conv weights ----
  k_pool_part<bf16_t><<<dim3(64, B2), 128, 0, stream>>>(combs, part, 128);
  k_pool_fin<<<B2, 128, 0, stream>>>(part, poolTop, 128);
  k_fc<<<1, B2*8, 0, stream>>>(poolTop, TCAW1, nullptr, topMid, 128, 8, 0);
  k_fc<<<1, B2*128, 0, stream>>>(topMid, TCAW2, nullptr, sTop, 8, 128, 2);
  k_scale_adjT<<<(B2*128*9*128+255)/256, 256, 0, stream>>>(adjT, sTop, adjTb);
  g_adj<<<1024, 128, 0, stream>>>(combs, adjTb, ADJB, X, OUT);
}
